// Moore2016_4148938408276
// MI455X (gfx1250) — matrix operand order repaired and checked by reading, not compiled, not run on hardware
//
#include <hip/hip_runtime.h>
#include <cstdint>
#include <cstddef>

// ---------------- CDNA5 WMMA types ----------------
typedef __attribute__((ext_vector_type(16))) __bf16 v16bf;
typedef __attribute__((ext_vector_type(8)))  float  v8f;

#define B_    16
#define C_    2
#define T_    320000
#define F_    625
#define HOP_  512
#define NCOMP 67
#define KPAD  96
#define NERB  150
#define NERBP 160
#define NROW  20000   // B*C*F

// component segments: {win start comp, N, first bin k, table float offset}
__device__ __constant__ int SEG_START[7] = {0,4,17,29,40,46,67};
__device__ __constant__ int SEG_N[6]     = {2048,1024,512,256,128,64};
__device__ __constant__ int SEG_K0[6]    = {2,3,8,10,11,9};
__device__ __constant__ int SEG_OFF[6]   = {0,16384,43008,55296,60928,62464};

static __device__ __forceinline__ unsigned short f2bf(float f) {
  unsigned u = __builtin_bit_cast(unsigned, f);
  u += 0x7FFFu + ((u >> 16) & 1u);          // round-to-nearest-even
  return (unsigned short)(u >> 16);
}

union V16 { v16bf v; uint4 q[2]; unsigned short s[16]; };

// ---------------- K0a: windowed-DFT coefficient table ----------------
__global__ void k_init_tab(float* __restrict__ tab) {
  int j = blockIdx.x;
  int w = 0;
  while (w < 5 && j >= SEG_START[w + 1]) ++w;
  int local = j - SEG_START[w];
  int N  = SEG_N[w];
  int kk = SEG_K0[w] + local;
  int off = SEG_OFF[w] + local * 2 * N;
  const float TWO_PI = 6.28318530717958647692f;
  for (int nn = threadIdx.x; nn < N; nn += blockDim.x) {
    float ang = TWO_PI * (float)(kk * nn) / (float)N;
    float win = 0.5f - 0.5f * cosf(TWO_PI * (float)nn / (float)(N - 1));
    tab[off + nn]     = win * cosf(ang);
    tab[off + N + nn] = win * sinf(ang);
  }
}

// ---------------- K0b: roex weights, transposed + padded, bf16 ----------------
// WT[e][k] = ROEX_W[k][e], e in [0,160), k in [0,96), zero padded
__global__ void k_init_roex(unsigned short* __restrict__ WT) {
  int idx = blockIdx.x * blockDim.x + threadIdx.x;
  if (idx >= NERBP * KPAD) return;
  int e = idx / KPAD, k = idx % KPAD;
  unsigned short out = 0;
  if (e < NERB && k < NCOMP) {
    int w = 0;
    while (w < 5 && k >= SEG_START[w + 1]) ++w;
    int kbin = SEG_K0[w] + (k - SEG_START[w]);
    float cf   = (float)kbin * 32000.f / (float)SEG_N[w];
    float erb  = 1.75f + 0.25f * (float)e;
    float fc   = (powf(10.f, erb / 21.366f) - 1.f) / 0.004368f;
    float erbw = 24.673f * (0.004368f * fc + 1.f);
    float p    = 4.f * fc / erbw;
    float gg   = fabsf(cf / fc - 1.f);
    out = f2bf((1.f + p * gg) * expf(-p * gg));
  }
  WT[idx] = out;
}

// ---------------- K1: ear FIR via Toeplitz bf16 WMMA ----------------
// y[t] = sum_j hr[j] * x[t-512+j],  hr[j] = h[1024-j]
// wave tile: C[m,n] = y[t0 + 16n + m]; 40 K-chunks of 32
__global__ void k_fir(const float* __restrict__ x, const float* __restrict__ fir,
                      float* __restrict__ y) {
  __shared__ __align__(16) unsigned short lds_h[1536];
  __shared__ __align__(16) unsigned short lds_x[8][1312];
  const int tid  = threadIdx.x;
  const int wv   = tid >> 5;
  const int lane = tid & 31;

  // reversed + front-padded (240 zeros) filter, bf16
  for (int i = tid; i < 1536; i += 256) {
    float hv = (i >= 240 && i < 1265) ? fir[1264 - i] : 0.f;
    lds_h[i] = f2bf(hv);
  }

  const int gtile = blockIdx.x * 8 + wv;     // 40000 tiles total
  const int s  = gtile / 1250;               // stream = b*2+c
  const int t0 = (gtile % 1250) * 256;
  const long base = (long)t0 - 512;
  const float* xs = x + (size_t)s * T_;
  for (int i = lane; i < 1312; i += 32) {
    long t = base + i;
    float v = (t >= 0 && t < T_) ? xs[t] : 0.f;
    lds_x[wv][i] = f2bf(v);
  }
  __syncthreads();

  const int gh = lane >> 4;                  // lane half
  const int m  = lane & 15;                  // A row / B col / D col
  const unsigned short* xw = lds_x[wv];
  v8f c = {};
  for (int kbase = 0; kbase < 1280; kbase += 32) {
    V16 a, b;
    const int a0 = m + kbase + 8 * gh;       // A: k = 8g + e (+8 for e>=8)
#pragma unroll
    for (int e = 0; e < 8; ++e)  a.s[e] = xw[a0 + e];
#pragma unroll
    for (int e = 8; e < 16; ++e) a.s[e] = xw[a0 + 8 + e];
    const int b0 = 240 + kbase + 8 * gh - 16 * m;   // B: k = 8g + e (+8 for e>=8) (16B aligned)
    const uint4* hp = (const uint4*)(&lds_h[b0]);
    b.q[0] = hp[0];
    b.q[1] = hp[2];
    c = __builtin_amdgcn_wmma_f32_16x16x32_bf16(false, a.v, false, b.v,
                                                (short)0, c, false, false);
  }
  // D: lane col n=m, rows r+8g -> 8 contiguous outputs
  float* yo = y + (size_t)s * T_ + t0 + 16 * m + 8 * gh;
  *(float4*)(yo)     = make_float4(c[0], c[1], c[2], c[3]);
  *(float4*)(yo + 4) = make_float4(c[4], c[5], c[6], c[7]);
}

// ---------------- K2: sparse spectrum (direct DFT on selected bins) ----------------
// Ipad bf16 [NROW][96], zero-padded K
__global__ void k_spectrum(const float* __restrict__ y, const float* __restrict__ tab,
                           unsigned short* __restrict__ Ipad) {
  __shared__ float fy[2048];
  const int row = blockIdx.x;                // bc*F + f
  const int f  = row % F_;
  const int bc = row / F_;
  const long b0 = (long)f * HOP_ - 1024;
  const float* ys = y + (size_t)bc * T_;
  for (int i = threadIdx.x; i < 2048; i += blockDim.x) {
    long t = b0 + i;
    fy[i] = (t >= 0 && t < T_) ? ys[t] : 0.f;
  }
  __syncthreads();
  const int j = threadIdx.x;
  if (j < NCOMP) {
    int w = 0;
    while (w < 5 && j >= SEG_START[w + 1]) ++w;
    int local = j - SEG_START[w];
    int N   = SEG_N[w];
    int off = SEG_OFF[w] + local * 2 * N;
    int o0  = 1024 - N / 2;
    float ac = 0.f, as = 0.f;
    for (int nn = 0; nn < N; ++nn) {
      float v = fy[o0 + nn];
      ac = fmaf(v, tab[off + nn], ac);
      as = fmaf(v, tab[off + N + nn], as);
    }
    float wsum = 0.5f * (float)(N - 1);       // Hann sum closed form
    float I = 2.f * (ac * ac + as * as) / (wsum * wsum);
    Ipad[(size_t)row * KPAD + j] = f2bf(I);
  } else if (j < KPAD) {
    Ipad[(size_t)row * KPAD + j] = 0;
  }
}

// ---------------- K3: E = I @ W (bf16 WMMA) + dB + LUT interp ----------------
__global__ void k_gemm(const unsigned short* __restrict__ Ipad,
                       const unsigned short* __restrict__ WT,
                       const float* __restrict__ lut,
                       float* __restrict__ nout) {
  const int wv = threadIdx.x >> 5, lane = threadIdx.x & 31;
  const int tile = blockIdx.x * 8 + wv;
  if (tile >= 12500) return;                 // 1250 M-tiles x 10 N-tiles
  const int mt = tile / 10, nt = tile % 10;
  const int r0 = mt * 16, n0 = nt * 16;
  const int gh = lane >> 4, ln = lane & 15;
  v8f c = {};
#pragma unroll
  for (int kb = 0; kb < KPAD; kb += 32) {
    V16 a, b;
    const unsigned short* ap = Ipad + (size_t)(r0 + ln) * KPAD + kb + 8 * gh;
    a.q[0] = *(const uint4*)(ap);            // e=0..7
    a.q[1] = *(const uint4*)(ap + 16);       // e=8..15 (k+16)
    const unsigned short* bp = WT + (size_t)(n0 + ln) * KPAD + kb + 8 * gh;
    b.q[0] = *(const uint4*)(bp);
    b.q[1] = *(const uint4*)(bp + 16);
    c = __builtin_amdgcn_wmma_f32_16x16x32_bf16(false, a.v, false, b.v,
                                                (short)0, c, false, false);
  }
  const int col = n0 + ln;
  if (col < NERB) {
    const float istep = 87.f / 110.f;        // uniform LUT grid
#pragma unroll
    for (int r = 0; r < 8; ++r) {
      int row = r0 + 8 * gh + r;
      float E  = c[r];
      float db = 10.f * log10f(E * 2.5e9f + 1e-12f);   // /I0, I0=4e-10
      float xi = fminf(fmaxf(db, 0.f), 110.f);
      float t  = xi * istep;
      int i0 = (int)t; if (i0 > 86) i0 = 86;
      float fr = t - (float)i0;
      nout[(size_t)row * NERB + col] = lut[i0] + fr * (lut[i0 + 1] - lut[i0]);
    }
  }
}

// ---------------- K4: short-term AGC (sequential over F, coalesced over erb) ----
__global__ void k_agc1(const float* __restrict__ nin, float* __restrict__ stl) {
  int tid = blockIdx.x * blockDim.x + threadIdx.x;
  if (tid >= 32 * NERB) return;
  int bc = tid / NERB, e = tid % NERB;
  float s = 0.f;
  const size_t base = (size_t)bc * F_ * NERB + e;
  for (int f = 0; f < F_; ++f) {
    float xv = nin[base + (size_t)f * NERB];
    float a = (xv > s) ? 0.045f : 0.033f;
    s = a * xv + (1.f - a) * s;
    stl[base + (size_t)f * NERB] = s;
  }
}

// ---------------- K5: 17-tap Gaussian smoothing along ERB axis ----------------
__global__ void k_smooth(const float* __restrict__ stl, float* __restrict__ sm) {
  int idx = blockIdx.x * blockDim.x + threadIdx.x;
  if (idx >= NROW * NERB) return;
  int row = idx / NERB, e = idx % NERB;
  float gw[17]; float gs = 0.f;
#pragma unroll
  for (int d = 0; d < 17; ++d) {
    float z = ((float)d - 8.f) * 0.25f;
    gw[d] = expf(-z * z / (2.f * 0.08f * 0.08f));
    gs += gw[d];
  }
  float acc = 0.f;
  const float* rp = stl + (size_t)row * NERB;
#pragma unroll
  for (int d = 0; d < 17; ++d) {
    int ee = e + d - 8;
    if (ee >= 0 && ee < NERB) acc += gw[d] * rp[ee];
  }
  sm[idx] = acc / gs;
}

// ---------------- K6: binaural inhibition + ERB sum ----------------
__global__ void k_binaural(const float* __restrict__ sm, float* __restrict__ stlr) {
  int idx = blockIdx.x * blockDim.x + threadIdx.x;
  if (idx >= B_ * F_) return;
  int b = idx / F_, f = idx % F_;
  const float* pl = sm + ((size_t)(b * 2 + 0) * F_ + f) * NERB;
  const float* pr = sm + ((size_t)(b * 2 + 1) * F_ + f) * NERB;
  float sl = 0.f, sr = 0.f;
  for (int e = 0; e < NERB; ++e) {
    float l = pl[e], r = pr[e];
    sl += l / coshf(1.5978f * r);
    sr += r / coshf(1.5978f * l);
  }
  stlr[idx]            = 0.25f * sl;
  stlr[B_ * F_ + idx]  = 0.25f * sr;
}

// ---------------- K7: long-term AGC + outputs ----------------
__global__ void k_final(const float* __restrict__ stlr, float* __restrict__ out) {
  int b = threadIdx.x;
  if (b >= B_) return;
  const float* L = stlr + (size_t)b * F_;
  const float* R = stlr + (size_t)(B_ * F_) + (size_t)b * F_;
  float sl = 0.f, sr = 0.f, mx = -1e30f;
  for (int f = 0; f < F_; ++f) {
    float xl = L[f], xr = R[f];
    out[b * F_ + f] = xl + xr;                 // sLoud
    float al = (xl > sl) ? 0.01f : 0.00133f;
    sl = al * xl + (1.f - al) * sl;
    float ar = (xr > sr) ? 0.01f : 0.00133f;
    sr = ar * xr + (1.f - ar) * sr;
    float ll = sl + sr;
    out[B_ * F_ + b * F_ + f] = ll;            // lLoud
    mx = fmaxf(mx, ll);
  }
  out[2 * B_ * F_ + b] = mx;                   // mLoud
}

// ---------------- workspace layout (bytes, 256-aligned) ----------------
static const size_t OFF_Y   = 0;                         // 40,960,000 f32
static const size_t OFF_TAB = 40960000;                  // 65,152 f32
static const size_t OFF_WT  = 41220608;                  // 160*96 bf16
static const size_t OFF_I   = 41251328;                  // 20000*96 bf16
static const size_t OFF_N   = 45091328;                  // 12,000,000 (n_inst; reused for sm)
static const size_t OFF_STL = 57091328;                  // 12,000,000
static const size_t OFF_LR  = 69091328;                  // 80,000
// total ~69.2 MB

extern "C" void kernel_launch(void* const* d_in, const int* in_sizes, int n_in,
                              void* d_out, int out_size, void* d_ws, size_t ws_size,
                              hipStream_t stream) {
  (void)in_sizes; (void)n_in; (void)out_size; (void)ws_size;
  const float* audio = (const float*)d_in[0];
  const float* fir   = (const float*)d_in[1];
  const float* lut   = (const float*)d_in[2];
  float* out = (float*)d_out;
  char* ws = (char*)d_ws;
  float* y             = (float*)(ws + OFF_Y);
  float* tab           = (float*)(ws + OFF_TAB);
  unsigned short* WT   = (unsigned short*)(ws + OFF_WT);
  unsigned short* Ipad = (unsigned short*)(ws + OFF_I);
  float* nbuf          = (float*)(ws + OFF_N);
  float* stl           = (float*)(ws + OFF_STL);
  float* sm            = (float*)(ws + OFF_N);  // overlay: nbuf dead after k_agc1
  float* stlr          = (float*)(ws + OFF_LR);

  k_init_tab <<<NCOMP, 256, 0, stream>>>(tab);
  k_init_roex<<<(NERBP * KPAD + 255) / 256, 256, 0, stream>>>(WT);
  k_fir      <<<5000, 256, 0, stream>>>(audio, fir, y);
  k_spectrum <<<NROW, 128, 0, stream>>>(y, tab, Ipad);
  k_gemm     <<<(12500 + 7) / 8, 256, 0, stream>>>(Ipad, WT, lut, nbuf);
  k_agc1     <<<(32 * NERB + 255) / 256, 256, 0, stream>>>(nbuf, stl);
  k_smooth   <<<(NROW * NERB + 255) / 256, 256, 0, stream>>>(stl, sm);
  k_binaural <<<(B_ * F_ + 255) / 256, 256, 0, stream>>>(sm, stlr);
  k_final    <<<1, 32, 0, stream>>>(stlr, out);
}
